// TransformerLayer_79688823210388
// MI455X (gfx1250) — matrix operand order repaired and checked by reading, not compiled, not run on hardware
//
#include <hip/hip_runtime.h>
#include <hip/hip_bf16.h>
#include <stdint.h>

// ---------------------------------------------------------------------------
// TransformerLayer for MI455X (gfx1250, wave32, WMMA bf16 16x16x32 + TDM)
// B=2, T=2048, D=2048, NH=16, HD=128
// ---------------------------------------------------------------------------

typedef __attribute__((ext_vector_type(16))) __bf16       v16bf;
typedef __attribute__((ext_vector_type(8)))  __bf16       v8bf;
typedef __attribute__((ext_vector_type(8)))  float        v8f;
typedef __attribute__((ext_vector_type(4)))  unsigned int u32x4;
typedef __attribute__((ext_vector_type(8)))  unsigned int u32x8;

union Frag16 {
  v16bf v;
  struct { v8bf lo, hi; } p;
};

constexpr int kB  = 2;
constexpr int kT  = 2048;
constexpr int kD  = 2048;
constexpr int kNH = 16;
constexpr int kHD = 128;

// ---------------------------------------------------------------------------
// 16-lane (half-wave) butterfly reductions via DPP16 — pure VALU, no LDS.
// Sequence quad_perm[1,0,3,2], quad_perm[2,3,0,1], row_half_mirror,
// row_mirror fully reduces each group of 16 lanes.
// ---------------------------------------------------------------------------
#if __has_builtin(__builtin_amdgcn_update_dpp)
template <int CTRL>
__device__ __forceinline__ float dpp_xchg(float x) {
  return __int_as_float(
      __builtin_amdgcn_update_dpp(0, __float_as_int(x), CTRL, 0xF, 0xF, true));
}
__device__ __forceinline__ float red_max16(float x) {
  x = fmaxf(x, dpp_xchg<0xB1>(x));   // quad_perm [1,0,3,2]
  x = fmaxf(x, dpp_xchg<0x4E>(x));   // quad_perm [2,3,0,1]
  x = fmaxf(x, dpp_xchg<0x141>(x));  // row_half_mirror
  x = fmaxf(x, dpp_xchg<0x140>(x));  // row_mirror
  return x;
}
__device__ __forceinline__ float red_sum16(float x) {
  x += dpp_xchg<0xB1>(x);
  x += dpp_xchg<0x4E>(x);
  x += dpp_xchg<0x141>(x);
  x += dpp_xchg<0x140>(x);
  return x;
}
#else
__device__ __forceinline__ float red_max16(float x) {
  for (int off = 1; off < 16; off <<= 1) x = fmaxf(x, __shfl_xor(x, off, 32));
  return x;
}
__device__ __forceinline__ float red_sum16(float x) {
  for (int off = 1; off < 16; off <<= 1) x += __shfl_xor(x, off, 32);
  return x;
}
#endif

// ---------------------------------------------------------------------------
// Tensor Data Mover: async DMA of a 2D bf16 tile (global -> LDS).
// D# group0: count=1, lds byte address, 57-bit global byte address, type=2.
// D# group1: data_size=2B, tensor_dim0/1, tile_dim0/1 (tile_dim2=0),
//            tensor_dim0_stride (elements). 2-group form => VADDR2/3 = NULL.
// Issued wave-uniform; tracked by TENSORcnt.
// ---------------------------------------------------------------------------
__device__ __forceinline__ void tdm_load_2d_bf16(unsigned lds_off,
                                                 const void* gsrc,
                                                 unsigned tensor_d0,
                                                 unsigned tensor_d1,
                                                 unsigned stride0_elems,
                                                 unsigned tile_d0,
                                                 unsigned tile_d1) {
  uint64_t ga = (uint64_t)(uintptr_t)gsrc;
  u32x4 g0;
  g0[0] = 1u;                                                 // count=1
  g0[1] = lds_off;                                            // lds_addr (bytes)
  g0[2] = (unsigned)ga;                                       // global_addr[31:0]
  g0[3] = ((unsigned)(ga >> 32) & 0x01FFFFFFu) | 0x80000000u; // addr[56:32]|type=2
  u32x8 g1;
  g1[0] = 1u << 16;                                           // data_size = 2 bytes
  g1[1] = (tensor_d0 & 0xFFFFu) << 16;
  g1[2] = (tensor_d0 >> 16) | ((tensor_d1 & 0xFFFFu) << 16);
  g1[3] = (tensor_d1 >> 16) | (tile_d0 << 16);
  g1[4] = tile_d1;                                            // tile_dim2 = 0
  g1[5] = stride0_elems;
  g1[6] = 0u;
  g1[7] = 0u;
  asm volatile("tensor_load_to_lds %0, %1" :: "s"(g0), "s"(g1) : "memory");
}

// ---------------------------------------------------------------------------
// fp32 -> bf16 cast (grid-stride)
// ---------------------------------------------------------------------------
__global__ void cast_f32_to_bf16(const float* __restrict__ src,
                                 __bf16* __restrict__ dst, size_t n) {
  size_t i = (size_t)blockIdx.x * blockDim.x + threadIdx.x;
  size_t stride = (size_t)gridDim.x * blockDim.x;
  for (; i < n; i += stride) dst[i] = (__bf16)src[i];
}

// ---------------------------------------------------------------------------
// bf16 WMMA GEMM with TDM double-buffered LDS staging.
// C[M,N] = A[M,K] * B[K,N]. Block = 128 threads = 4 waves, tile 64x64.
// K-slab 64 deep: TDM loads A (64x64) and B (64x64) slabs into LDS while the
// previous slab is consumed. Dynamic LDS: A[2]@{0,8192}, B[2]@{16384,24576}.
// ---------------------------------------------------------------------------
template<bool OUT_BF16>
__global__ __launch_bounds__(128)
void gemm_bf16_wmma(const __bf16* __restrict__ A, const __bf16* __restrict__ Bw,
                    void* __restrict__ C, int M, int N, int K) {
  extern __shared__ char smem[];
  const int wave = threadIdx.x >> 5;
  const int lane = threadIdx.x & 31;
  const int h    = lane >> 4;
  const int lr   = lane & 15;
  const int mblk = blockIdx.y * 64;
  const int n0   = blockIdx.x * 64;

  v8f acc[4] = {v8f{}, v8f{}, v8f{}, v8f{}};
  const int nk = K >> 6;

  if (wave == 0) {
    tdm_load_2d_bf16(0u, A + (size_t)mblk * K,
                     (unsigned)K, (unsigned)M, (unsigned)K, 64u, 64u);
    tdm_load_2d_bf16(16384u, Bw + n0,
                     (unsigned)N, (unsigned)K, (unsigned)N, 64u, 64u);
  }

  for (int i = 0; i < nk; ++i) {
    const unsigned cur = (unsigned)(i & 1);
    const unsigned aOff = cur * 8192u;            // 0 / 8192
    const unsigned bOff = 16384u + cur * 8192u;   // 16384 / 24576
    if (wave == 0) {
      if (i + 1 < nk) {
        const unsigned nxt = cur ^ 1u;
        tdm_load_2d_bf16(nxt * 8192u, A + (size_t)mblk * K + (i + 1) * 64,
                         (unsigned)K, (unsigned)M, (unsigned)K, 64u, 64u);
        tdm_load_2d_bf16(16384u + nxt * 8192u,
                         Bw + (size_t)(i + 1) * 64 * N + n0,
                         (unsigned)N, (unsigned)K, (unsigned)N, 64u, 64u);
        __builtin_amdgcn_s_wait_tensorcnt((short)2);  // current slab landed
      } else {
        __builtin_amdgcn_s_wait_tensorcnt((short)0);
      }
    }
    __syncthreads();

    const __bf16* As = (const __bf16*)(smem + aOff);
    const __bf16* Bs = (const __bf16*)(smem + bOff);
#pragma unroll
    for (int s2 = 0; s2 < 2; ++s2) {
      Frag16 a;
      const __bf16* ar = As + (size_t)(wave * 16 + lr) * 64 + s2 * 32;
      a.p.lo = *(const v8bf*)(ar + 8 * h);
      a.p.hi = *(const v8bf*)(ar + 16 + 8 * h);

      const __bf16* br = Bs + (size_t)(s2 * 32) * 64 + lr;
#pragma unroll
      for (int j = 0; j < 4; ++j) {
        v16bf b;
#pragma unroll
        for (int s = 0; s < 16; ++s)
          b[s] = br[(size_t)((s < 8) ? (8 * h + s) : (8 + 8 * h + s)) * 64 + j * 16];
        acc[j] = __builtin_amdgcn_wmma_f32_16x16x32_bf16(
            false, a.v, false, b, (short)0, acc[j], false, false);
      }
    }
    __syncthreads();  // everyone done reading before TDM overwrites this buffer
  }

  const int row0 = mblk + wave * 16 + 8 * h;
#pragma unroll
  for (int i = 0; i < 8; ++i) {
    size_t base = (size_t)(row0 + i) * N + n0 + lr;
    if constexpr (OUT_BF16) {
      __bf16* Cp = (__bf16*)C;
#pragma unroll
      for (int j = 0; j < 4; ++j) Cp[base + j * 16] = (__bf16)acc[j][i];
    } else {
      float* Cp = (float*)C;
#pragma unroll
      for (int j = 0; j < 4; ++j) Cp[base + j * 16] = acc[j][i];
    }
  }
}

// ---------------------------------------------------------------------------
// RoPE + head transpose.
// qkv: bf16 [B*T, 3*D] -> qb [B,NH,T,HD], kbt [B,NH,HD,T] (transposed),
// vb [B,NH,T,HD]. One thread per (b,h,t,i), i in [0,64).
// ---------------------------------------------------------------------------
__global__ __launch_bounds__(256)
void rope_transpose(const __bf16* __restrict__ qkv,
                    __bf16* __restrict__ qb,
                    __bf16* __restrict__ kbt,
                    __bf16* __restrict__ vb) {
  size_t gid = (size_t)blockIdx.x * 256 + threadIdx.x;
  int    i   = (int)(gid & 63);
  size_t r   = gid >> 6;
  int    t   = (int)(r % kT);
  size_t bh  = r / kT;
  int    head = (int)(bh % kNH);

  const __bf16* src = qkv + (size_t)(bh / kNH) * kT * (3 * kD) + (size_t)t * (3 * kD);
  int c = head * kHD;

  float q1 = (float)src[c + i],          q2 = (float)src[c + 64 + i];
  float k1 = (float)src[kD + c + i],     k2 = (float)src[kD + c + 64 + i];
  float v1 = (float)src[2 * kD + c + i], v2 = (float)src[2 * kD + c + 64 + i];

  float freq = __expf(-(float)i * (9.210340371976184f / 64.0f));
  float ang  = (float)t * freq;
  float cs = __cosf(ang);
  float sn = __sinf(ang);

  size_t o = (bh * kT + t) * kHD;
  qb[o + i]      = (__bf16)(q1 * cs - q2 * sn);
  qb[o + 64 + i] = (__bf16)(q2 * cs + q1 * sn);

  size_t ot = bh * (size_t)kHD * kT;
  kbt[ot + (size_t)i * kT + t]        = (__bf16)(k1 * cs - k2 * sn);
  kbt[ot + (size_t)(i + 64) * kT + t] = (__bf16)(k2 * cs + k1 * sn);

  vb[o + i]      = (__bf16)v1;
  vb[o + 64 + i] = (__bf16)v2;
}

// ---------------------------------------------------------------------------
// Flash-style causal attention with TDM-staged K/V tiles.
// Block = 4 waves = 64 query rows; each wave owns 16 rows.
// Per 32-key block: wave 0 TDM-loads K tile (128hd x 32t from kbt) and
// V tile (32t x 128d) into double-buffered LDS shared by all 4 waves.
// Online softmax in fp32 registers (DPP16 butterflies for row max/sum);
// P round-trips a per-wave LDS tile to get the A-fragment layout for P @ V.
// Dynamic LDS: K[2]@{0,8192}, V[2]@{16384,24576}, P@32768+wave*1024 (36 KB)
// ---------------------------------------------------------------------------
__global__ __launch_bounds__(128)
void attn_flash_wmma(const __bf16* __restrict__ qb,
                     const __bf16* __restrict__ kbt,
                     const __bf16* __restrict__ vb,
                     __bf16* __restrict__ ao) {
  extern __shared__ char smem[];
  const int wave = threadIdx.x >> 5;
  const int lane = threadIdx.x & 31;
  const int h    = lane >> 4;
  const int lr   = lane & 15;

  const int qt   = blockIdx.x;
  const int head = blockIdx.y;
  const int b    = blockIdx.z;

  const size_t bh = (size_t)b * kNH + head;
  const __bf16* Q  = qb  + bh * (size_t)kT * kHD;
  const __bf16* Kt = kbt + bh * (size_t)kHD * kT;
  const __bf16* V  = vb  + bh * (size_t)kT * kHD;

  __bf16* Ps = (__bf16*)(smem + 32768 + wave * 1024);  // [16][32]

  const int q0 = qt * 64 + wave * 16;

  // Q fragments: 4 k-steps of 32 over HD=128 (held in registers)
  Frag16 qf[4];
  const __bf16* qrow = Q + (size_t)(q0 + lr) * kHD;
#pragma unroll
  for (int s = 0; s < 4; ++s) {
    qf[s].p.lo = *(const v8bf*)(qrow + s * 32 + 8 * h);
    qf[s].p.hi = *(const v8bf*)(qrow + s * 32 + 16 + 8 * h);
  }

  float m_i[8], l_i[8];
  v8f acc[8];
#pragma unroll
  for (int i = 0; i < 8; ++i) { m_i[i] = -3.0e38f; l_i[i] = 0.0f; acc[i] = v8f{}; }

  const float scale = 0.08838834764831845f;  // 1/sqrt(128)
  const int nkb = (qt + 1) * 2;              // causal: 32-key blocks, uniform

  if (wave == 0) {
    tdm_load_2d_bf16(0u, Kt, (unsigned)kT, (unsigned)kHD, (unsigned)kT,
                     32u, 128u);
    tdm_load_2d_bf16(16384u, V, (unsigned)kHD, (unsigned)kT, (unsigned)kHD,
                     128u, 32u);
  }

  for (int kb = 0; kb < nkb; ++kb) {
    const unsigned cur = (unsigned)(kb & 1);
    const int k0 = kb * 32;
    if (wave == 0) {
      if (kb + 1 < nkb) {
        const unsigned nxt = cur ^ 1u;
        tdm_load_2d_bf16(nxt * 8192u, Kt + (kb + 1) * 32,
                         (unsigned)kT, (unsigned)kHD, (unsigned)kT, 32u, 128u);
        tdm_load_2d_bf16(16384u + nxt * 8192u, V + (size_t)(kb + 1) * 32 * kHD,
                         (unsigned)kHD, (unsigned)kT, (unsigned)kHD, 128u, 32u);
        __builtin_amdgcn_s_wait_tensorcnt((short)2);
      } else {
        __builtin_amdgcn_s_wait_tensorcnt((short)0);
      }
    }
    __syncthreads();

    const __bf16* Ks = (const __bf16*)(smem + cur * 8192u);            // [128][32]
    const __bf16* Vs = (const __bf16*)(smem + 16384u + cur * 8192u);   // [32][128]

    // ---- scores S = Q K^T (two 16x16 tiles over this 32-key block) ----
    v8f s0 = v8f{}, s1 = v8f{};
#pragma unroll
    for (int st = 0; st < 4; ++st) {
      const __bf16* krow = Ks + (size_t)(st * 32) * 32 + lr;
      v16bf bk0, bk1;
#pragma unroll
      for (int s = 0; s < 16; ++s) {
        const int kk = (s < 8) ? (8 * h + s) : (8 + 8 * h + s);
        bk0[s] = krow[kk * 32];
        bk1[s] = krow[kk * 32 + 16];
      }
      s0 = __builtin_amdgcn_wmma_f32_16x16x32_bf16(false, qf[st].v, false, bk0,
                                                   (short)0, s0, false, false);
      s1 = __builtin_amdgcn_wmma_f32_16x16x32_bf16(false, qf[st].v, false, bk1,
                                                   (short)0, s1, false, false);
    }

    // ---- scale + causal mask + online softmax (row = q0 + i + 8h) ----
#pragma unroll
    for (int i = 0; i < 8; ++i) {
      const int qg = q0 + i + 8 * h;
      float e0 = s0[i] * scale;
      float e1 = s1[i] * scale;
      e0 = (k0 + lr      <= qg) ? e0 : -3.0e38f;
      e1 = (k0 + 16 + lr <= qg) ? e1 : -3.0e38f;

      const float mx   = red_max16(fmaxf(e0, e1));
      const float mnew = fmaxf(m_i[i], mx);
      const float sc   = __expf(m_i[i] - mnew);
      m_i[i] = mnew;

      const float p0 = __expf(e0 - mnew);
      const float p1 = __expf(e1 - mnew);
      l_i[i] = l_i[i] * sc + red_sum16(p0 + p1);

#pragma unroll
      for (int n = 0; n < 8; ++n) acc[n][i] *= sc;

      s0[i] = p0;
      s1[i] = p1;
    }

    // ---- C-layout P -> per-wave LDS tile -> A-fragment layout ----
#pragma unroll
    for (int i = 0; i < 8; ++i) {
      Ps[(i + 8 * h) * 32 + lr]      = (__bf16)s0[i];
      Ps[(i + 8 * h) * 32 + 16 + lr] = (__bf16)s1[i];
    }
    asm volatile("s_wait_dscnt 0" ::: "memory");

    Frag16 pf;
    pf.p.lo = *(const v8bf*)(Ps + lr * 32 + 8 * h);
    pf.p.hi = *(const v8bf*)(Ps + lr * 32 + 16 + 8 * h);

    // ---- acc += P @ V (8 column tiles over HD=128) ----
    const __bf16* vrow = Vs + lr;
#pragma unroll
    for (int n = 0; n < 8; ++n) {
      v16bf bv;
#pragma unroll
      for (int s = 0; s < 16; ++s)
        bv[s] = vrow[((s < 8) ? (8 * h + s) : (8 + 8 * h + s)) * kHD + n * 16];
      acc[n] = __builtin_amdgcn_wmma_f32_16x16x32_bf16(false, pf.v, false, bv,
                                                       (short)0, acc[n], false, false);
    }
    __syncthreads();  // done reading this K/V buffer before TDM refills it
  }

  // ---- epilogue: normalize and write bf16 [B,T,D] ----
#pragma unroll
  for (int i = 0; i < 8; ++i) {
    const float inv = 1.0f / l_i[i];
    const int trow = q0 + i + 8 * h;
    size_t base = ((size_t)b * kT + trow) * kD + (size_t)head * kHD + lr;
#pragma unroll
    for (int n = 0; n < 8; ++n)
      ao[base + n * 16] = (__bf16)(acc[n][i] * inv);
  }
}

// ---------------------------------------------------------------------------
// Launch
// ---------------------------------------------------------------------------
extern "C" void kernel_launch(void* const* d_in, const int* in_sizes, int n_in,
                              void* d_out, int out_size, void* d_ws, size_t ws_size,
                              hipStream_t stream) {
  const float* x     = (const float*)d_in[0];  // [B,T,D]
  const float* w_qkv = (const float*)d_in[1];  // [D, 3D]
  const float* w_o   = (const float*)d_in[2];  // [D, D]
  float* out = (float*)d_out;                  // [B,T,D] fp32

  const size_t BT = (size_t)kB * kT;           // 4096 rows

  char* w = (char*)d_ws;
  __bf16* xb    = (__bf16*)w; w += BT * kD * 2;                 // 16 MB
  __bf16* wqkvb = (__bf16*)w; w += (size_t)kD * 3 * kD * 2;     // 24 MB
  __bf16* wob   = (__bf16*)w; w += (size_t)kD * kD * 2;         //  8 MB
  __bf16* qkv   = (__bf16*)w; w += BT * 3 * kD * 2;             // 48 MB
  __bf16* qb    = (__bf16*)w; w += BT * kD * 2;                 // 16 MB
  __bf16* kbt   = (__bf16*)w; w += BT * kD * 2;                 // 16 MB
  __bf16* vb    = (__bf16*)w; w += BT * kD * 2;                 // 16 MB
  __bf16* ao    = (__bf16*)w; w += BT * kD * 2;                 // 16 MB

  // 1) cast inputs to bf16
  cast_f32_to_bf16<<<4096, 256, 0, stream>>>(x, xb, BT * kD);
  cast_f32_to_bf16<<<4096, 256, 0, stream>>>(w_qkv, wqkvb, (size_t)kD * 3 * kD);
  cast_f32_to_bf16<<<2048, 256, 0, stream>>>(w_o, wob, (size_t)kD * kD);

  // 2) qkv = x @ w_qkv : M=4096, N=6144, K=2048, bf16 out
  {
    dim3 grid(3 * kD / 64, BT / 64);  // (96, 64)
    gemm_bf16_wmma<true><<<grid, 128, 32768, stream>>>(xb, wqkvb, qkv,
                                                       (int)BT, 3 * kD, kD);
  }

  // 3) RoPE + head transpose
  {
    size_t total = (size_t)kB * kNH * kT * 64;
    rope_transpose<<<(unsigned)(total / 256), 256, 0, stream>>>(qkv, qb, kbt, vb);
  }

  // 4) causal flash attention (TDM-staged K/V)
  {
    dim3 grid(kT / 64, kNH, kB);  // (32, 16, 2)
    attn_flash_wmma<<<grid, 128, 36864, stream>>>(qb, kbt, vb, ao);
  }

  // 5) out = ao @ w_o : M=4096, N=2048, K=2048, fp32 out
  {
    dim3 grid(kD / 64, BT / 64);  // (32, 64)
    gemm_bf16_wmma<false><<<grid, 128, 32768, stream>>>(ao, wob, out,
                                                        (int)BT, kD, kD);
  }
}
